// CustomModel_57303453663822
// MI455X (gfx1250) — matrix operand order repaired and checked by reading, not compiled, not run on hardware
//
#include <hip/hip_runtime.h>

typedef __attribute__((ext_vector_type(16))) _Float16 v16h;
typedef __attribute__((ext_vector_type(8)))  _Float16 v8h;
typedef __attribute__((ext_vector_type(8)))  float    v8f;
typedef __attribute__((ext_vector_type(4)))  int      v4i;

#define B_SZ   256
#define T_SZ   512
#define EMB_D  256
#define UNITS  512
#define KTOT   768            // EMB_D + UNITS
#define NGATE  2048           // 4*UNITS, column index cz = gate*512 + unit
#define NWG    64             // 4 batch tiles x 16 unit groups
#define MT     64             // batch rows per WG
#define UT     32             // units per WG (gate cols per WG = 4*UT = 128)

#define HAVE_ASYNC_LDS __has_builtin(__builtin_amdgcn_global_load_async_to_lds_b128)

__device__ __forceinline__ float sigmoidf_(float x) { return 1.0f / (1.0f + __expf(-x)); }

__device__ __forceinline__ void wait_asynccnt0() {
#if __has_builtin(__builtin_amdgcn_s_wait_asynccnt)
  __builtin_amdgcn_s_wait_asynccnt(0);
#else
  asm volatile("s_wait_asynccnt 0x0" ::: "memory");
#endif
}

// ---------------------------------------------------------------------------
// Pack [W;U] for the 4 gates into Z^T[cz][k] (f16, K contiguous) + bias4[cz].
// cz = g*512 + u ; k < 256 -> W_g[k][u], k >= 256 -> U_g[k-256][u]
// ---------------------------------------------------------------------------
__global__ void pack_weights_kernel(
    const float* __restrict__ Wf, const float* __restrict__ Uf,
    const float* __restrict__ Wi, const float* __restrict__ Ui,
    const float* __restrict__ Wc, const float* __restrict__ Uc,
    const float* __restrict__ Wo, const float* __restrict__ Uo,
    const float* __restrict__ bf, const float* __restrict__ bi,
    const float* __restrict__ bc, const float* __restrict__ bo,
    _Float16* __restrict__ Zt, float* __restrict__ bias4)
{
  int idx = blockIdx.x * 256 + threadIdx.x;
  if (idx >= NGATE * KTOT) return;
  int cz = idx / KTOT;
  int k  = idx % KTOT;
  int g  = cz >> 9;          // /512
  int u  = cz & 511;
  const float* Wp = (g == 0) ? Wf : (g == 1) ? Wi : (g == 2) ? Wc : Wo;
  const float* Up = (g == 0) ? Uf : (g == 1) ? Ui : (g == 2) ? Uc : Uo;
  float v = (k < EMB_D) ? Wp[k * UNITS + u] : Up[(k - EMB_D) * UNITS + u];
  Zt[(size_t)cz * KTOT + k] = (_Float16)v;
  if (k == 0) {
    const float* bp = (g == 0) ? bf : (g == 1) ? bi : (g == 2) ? bc : bo;
    bias4[cz] = bp[u];
  }
}

__global__ void init_state_kernel(_Float16* __restrict__ hbuf, unsigned* __restrict__ cnt)
{
  int i = blockIdx.x * 256 + threadIdx.x;
  if (i < B_SZ * UNITS) hbuf[i] = (_Float16)0.0f;
  if (i == 0) *cnt = 0u;
}

// ---------------------------------------------------------------------------
// Persistent LSTM. Grid = 64 WGs x 256 threads (8 waves). Each WG owns
// rows [mblk*64, +64) x units [ug*32, +32). Per step: GEMM [64x768]x[768x128]
// via v_wmma_f32_16x16x32_f16 (96 WMMA/wave/step, 768/WG/step), then cell update with c in
// VGPRs, h -> global f16, grid-wide atomic barrier. h staging uses the
// gfx1250 async global->LDS path (ASYNCcnt) when available.
// ---------------------------------------------------------------------------
__global__ __launch_bounds__(256, 1) void lstm_persistent_kernel(
    const int*      __restrict__ tokens,
    const float*    __restrict__ emb,
    const _Float16* __restrict__ Zt,
    const float*    __restrict__ bias4,
    _Float16*       __restrict__ hbuf,
    unsigned*                    cnt)
{
  // A-staging (f16 [64][264], 33792 B) aliased with gate buffer (f32 [64][132])
  __shared__ __align__(16) unsigned char smraw[64 * 264 * 2];
  _Float16 (*As)[264] = reinterpret_cast<_Float16(*)[264]>(smraw);
  float    (*Gs)[132] = reinterpret_cast<float(*)[132]>(smraw);

  const int wg      = blockIdx.x;
  const int mblk    = wg >> 4;
  const int ug      = wg & 15;
  const int rowbase = mblk * MT;
  const int ubase   = ug * UT;

  const int tid  = threadIdx.x;
  const int wave = tid >> 5;
  const int lane = tid & 31;
  const int wm   = wave & 1;    // M half (rows 0..31 / 32..63)
  const int wn   = wave >> 1;   // gate index 0..3
  const int l15  = lane & 15;
  const int lhi  = lane >> 4;

  // staging: thread -> (row, 64-col chunk)
  const int sr = tid >> 2;
  const int sc = (tid & 3) * 64;
  // elementwise: thread -> (row, 8 units)
  const int er  = tid >> 2;
  const int eu0 = (tid & 3) * 8;

  float cst[8];
#pragma unroll
  for (int j = 0; j < 8; ++j) cst[j] = 0.0f;

  for (int t = 0; t < T_SZ; ++t) {
    v8f acc[2][2] = {};

#pragma unroll 1
    for (int kc = 0; kc < 3; ++kc) {
      __syncthreads();
      if (kc == 0) {
        // embedding lookup -> f16 A rows (K 0..255); needs f32->f16, VALU path
        int tok = tokens[(rowbase + sr) * T_SZ + t];
        const float* ep = emb + (size_t)tok * EMB_D + sc;
#pragma unroll
        for (int e = 0; e < 64; e += 4) {
          float4 f4 = *(const float4*)(ep + e);
          As[sr][sc + e + 0] = (_Float16)f4.x;
          As[sr][sc + e + 1] = (_Float16)f4.y;
          As[sr][sc + e + 2] = (_Float16)f4.z;
          As[sr][sc + e + 3] = (_Float16)f4.w;
        }
        if (t + 1 < T_SZ) {   // warm L2 for next step's embedding rows
          int tok2 = tokens[(rowbase + sr) * T_SZ + t + 1];
          __builtin_prefetch(emb + (size_t)tok2 * EMB_D + sc, 0, 1);
        }
      } else {
        // h_{t-1} (K 256..767), f16 in L2: async DMA straight into LDS
        const _Float16* hp = hbuf + (size_t)(rowbase + sr) * UNITS + (kc - 1) * 256 + sc;
#if HAVE_ASYNC_LDS
#pragma unroll
        for (int e = 0; e < 64; e += 8) {
          __builtin_amdgcn_global_load_async_to_lds_b128(
              (__attribute__((address_space(1))) v4i*)(hp + e),
              (__attribute__((address_space(3))) v4i*)(&As[sr][sc + e]),
              0, 0);
        }
        wait_asynccnt0();
#else
#pragma unroll
        for (int e = 0; e < 64; e += 8)
          *(v8h*)(&As[sr][sc + e]) = *(const v8h*)(hp + e);
#endif
      }
      __syncthreads();

#pragma unroll
      for (int kk = 0; kk < 8; ++kk) {
        // A fragments (contiguous k order used here, same as B: lanes 0-15 K 0..15,
        // lanes 16-31 K 16..31)
        v16h afr[2];
#pragma unroll
        for (int mi = 0; mi < 2; ++mi) {
          const _Float16* ap = &As[wm * 32 + mi * 16 + l15][kk * 32 + lhi * 16];
          v8h lo = *(const v8h*)ap;
          v8h hi = *(const v8h*)(ap + 8);
          afr[mi] = __builtin_shufflevector(lo, hi,
                      0, 1, 2, 3, 4, 5, 6, 7, 8, 9, 10, 11, 12, 13, 14, 15);
        }
        // B fragments straight from L2-resident Z^T: lane = column, 16
        // contiguous K per lane (lanes 0-15 K 0..15, lanes 16-31 K 16..31)
        v16h bfr[2];
#pragma unroll
        for (int ni = 0; ni < 2; ++ni) {
          int cz = wn * UNITS + ubase + ni * 16 + l15;
          bfr[ni] = *(const v16h*)(Zt + (size_t)cz * KTOT + kc * 256 + kk * 32 + lhi * 16);
        }
#pragma unroll
        for (int mi = 0; mi < 2; ++mi)
#pragma unroll
          for (int ni = 0; ni < 2; ++ni)
            acc[mi][ni] = __builtin_amdgcn_wmma_f32_16x16x32_f16(
                false, afr[mi], false, bfr[ni], (short)0, acc[mi][ni], false, false);
      }
    }

    __syncthreads();
    // C layout: lane l col N=l&15, VGPR v row M=v+(l>>4)*8
#pragma unroll
    for (int mi = 0; mi < 2; ++mi)
#pragma unroll
      for (int ni = 0; ni < 2; ++ni)
#pragma unroll
        for (int v = 0; v < 8; ++v)
          Gs[wm * 32 + mi * 16 + v + lhi * 8][wn * 32 + ni * 16 + l15] = acc[mi][ni][v];
    __syncthreads();

    // LSTM cell update; c stays in VGPRs across all 512 steps
#pragma unroll
    for (int j = 0; j < 8; ++j) {
      int u   = eu0 + j;
      int ugl = ubase + u;
      float f  = sigmoidf_(Gs[er][u]       + bias4[ugl]);
      float ii = sigmoidf_(Gs[er][32 + u]  + bias4[512 + ugl]);
      float ch = tanhf    (Gs[er][64 + u]  + bias4[1024 + ugl]);
      float o  = sigmoidf_(Gs[er][96 + u]  + bias4[1536 + ugl]);
      cst[j] = f * cst[j] + ii * ch;
      hbuf[(size_t)(rowbase + er) * UNITS + ugl] = (_Float16)(o * tanhf(cst[j]));
    }

    // grid-wide barrier (monotonic counter, zeroed by init_state each launch)
    __syncthreads();
    if (tid == 0) {
      __hip_atomic_fetch_add(cnt, 1u, __ATOMIC_RELEASE, __HIP_MEMORY_SCOPE_AGENT);
      unsigned target = (unsigned)NWG * (unsigned)(t + 1);
      while (__hip_atomic_load(cnt, __ATOMIC_ACQUIRE, __HIP_MEMORY_SCOPE_AGENT) < target) {
        __builtin_amdgcn_s_sleep(1);
      }
    }
    __syncthreads();
  }
}

// ---------------------------------------------------------------------------
// out[b] = sigmoid(h_T[b,:] . W + b0)
// ---------------------------------------------------------------------------
__global__ void head_kernel(const _Float16* __restrict__ hbuf,
                            const float* __restrict__ W,
                            const float* __restrict__ b0,
                            float* __restrict__ out)
{
  int bi = blockIdx.x * blockDim.x + threadIdx.x;
  if (bi >= B_SZ) return;
  float s = 0.0f;
#pragma unroll 8
  for (int u = 0; u < UNITS; ++u)
    s += (float)hbuf[(size_t)bi * UNITS + u] * W[u];
  out[bi] = sigmoidf_(s + b0[0]);
}

extern "C" void kernel_launch(void* const* d_in, const int* in_sizes, int n_in,
                              void* d_out, int out_size, void* d_ws, size_t ws_size,
                              hipStream_t stream)
{
  const int*   tokens = (const int*)d_in[0];
  const float* emb    = (const float*)d_in[1];
  const float* Wf = (const float*)d_in[2];  const float* Uf = (const float*)d_in[3];
  const float* bf = (const float*)d_in[4];
  const float* Wi = (const float*)d_in[5];  const float* Ui = (const float*)d_in[6];
  const float* bi = (const float*)d_in[7];
  const float* Wc = (const float*)d_in[8];  const float* Uc = (const float*)d_in[9];
  const float* bc = (const float*)d_in[10];
  const float* Wo = (const float*)d_in[11]; const float* Uo = (const float*)d_in[12];
  const float* bo = (const float*)d_in[13];
  const float* W  = (const float*)d_in[14]; const float* b0 = (const float*)d_in[15];

  // workspace layout
  const size_t Z_BYTES    = (size_t)NGATE * KTOT * sizeof(_Float16); // 3,145,728
  const size_t BIAS_BYTES = (size_t)NGATE * sizeof(float);           //     8,192
  const size_t H_BYTES    = (size_t)B_SZ * UNITS * sizeof(_Float16); //   262,144
  char* ws = (char*)d_ws;
  _Float16* Zt    = (_Float16*)ws;
  float*    bias4 = (float*)(ws + Z_BYTES);
  _Float16* hbuf  = (_Float16*)(ws + Z_BYTES + BIAS_BYTES);
  unsigned* cnt   = (unsigned*)(ws + Z_BYTES + BIAS_BYTES + H_BYTES);
  (void)in_sizes; (void)n_in; (void)out_size; (void)ws_size;

  pack_weights_kernel<<<(NGATE * KTOT + 255) / 256, 256, 0, stream>>>(
      Wf, Uf, Wi, Ui, Wc, Uc, Wo, Uo, bf, bi, bc, bo, Zt, bias4);
  init_state_kernel<<<(B_SZ * UNITS + 255) / 256, 256, 0, stream>>>(hbuf, cnt);
  lstm_persistent_kernel<<<NWG, 256, 0, stream>>>(tokens, emb, Zt, bias4, hbuf, cnt);
  head_kernel<<<1, 256, 0, stream>>>(hbuf, W, b0, (float*)d_out);
}
